// AttentionHead_6794638262727
// MI455X (gfx1250) — matrix operand order repaired and checked by reading, not compiled, not run on hardware
//
#include <hip/hip_runtime.h>
#include <hip/hip_bf16.h>

// ---------------------------------------------------------------------------
// AttentionHead on MI455X (gfx1250): f16 WMMA (v_wmma_f32_16x16x32_f16) for
// all five GEMMs, fp32 accumulate. Softmax is over the BATCH axis, so scores
// [64,512,512] (67MB fp32) are materialized in workspace (L2-resident, 192MB).
// Attn is re-emitted as f16 so the attn@V GEMM loads fragments directly.
// ---------------------------------------------------------------------------

typedef __attribute__((ext_vector_type(16))) _Float16 v16h;
typedef __attribute__((ext_vector_type(8)))  float    v8f;

union Frag16 {
    v16h v;
    _Float16 h[16];
    uint4 q[2];
};

#define BB   64
#define NN   512
#define EE   256
#define XPAD 264   // halves per LDS row of x strip (528B = 33*16B, conflict pad)
#define OPAD 260   // floats per LDS row of projection accumulator

// ---------------------------------------------------------------------------
// Kernel 0: one-time f32 -> f16 conversion of the three 256x256 weights.
// ---------------------------------------------------------------------------
__global__ __launch_bounds__(256) void cvt_weights_kernel(
    const float* __restrict__ Wq, const float* __restrict__ Wk,
    const float* __restrict__ Wv,
    _Float16* __restrict__ Wqh, _Float16* __restrict__ Wkh,
    _Float16* __restrict__ Wvh)
{
    const int i = blockIdx.x * 256 + threadIdx.x;   // 0 .. 65535
    Wqh[i] = (_Float16)Wq[i];
    Wkh[i] = (_Float16)Wk[i];
    Wvh[i] = (_Float16)Wv[i];
}

// ---------------------------------------------------------------------------
// Kernel 1: Q/K/V projection + bias + LayerNorm.
// Block = 256 threads (8 waves); one 16-row strip of one batch, all three
// 16x256 projected strips. Each wave owns 32 output columns (2 WMMA tiles,
// independent accumulators sharing one A-fragment).
// A-frag (x) from LDS (ds_load_b128); B-frag = one contiguous v16h of Wh row.
// Q,K stored row-major f16; V stored TRANSPOSED (Vt[b][e][m]) so kernel 4's
// B-fragments are contiguous.
// ---------------------------------------------------------------------------
__global__ __launch_bounds__(256) void qkv_ln_kernel(
    const float* __restrict__ x,
    const _Float16* __restrict__ Wqh, const float* __restrict__ bq,
    const _Float16* __restrict__ Wkh, const float* __restrict__ bk,
    const _Float16* __restrict__ Wvh, const float* __restrict__ bv,
    _Float16* __restrict__ Qh, _Float16* __restrict__ Kh,
    _Float16* __restrict__ Vt)
{
    __shared__ __align__(16) _Float16 xs[16 * XPAD];
    __shared__ __align__(16) float oacc[16 * OPAD];
    __shared__ float ps[16][16], ps2[16][16];
    __shared__ float mu_s[16], rs_s[16];

    const int b    = blockIdx.x;        // 64
    const int n0   = blockIdx.y * 16;   // 32 strips
    const int tid  = threadIdx.x;
    const int lane = tid & 31;
    const int wave = tid >> 5;
    const int lh   = lane >> 4;         // half-wave: A and B both get K+16 group
    const int lm   = lane & 15;         // M (A) or N (B/D) index

    // Stage x strip [16][256] as f16 into LDS.
    const float* xrow = x + ((size_t)b * NN + n0) * EE;
    for (int i = tid; i < 16 * EE; i += 256) {
        int r = i >> 8, c = i & 255;
        xs[r * XPAD + c] = (_Float16)xrow[r * EE + c];
    }
    __syncthreads();

    const _Float16* Ws[3] = {Wqh, Wkh, Wvh};
    const float*    bs[3] = {bq, bk, bv};

    for (int p = 0; p < 3; ++p) {
        const _Float16* W = Ws[p];
        const int d0 = wave * 32;       // this wave's 32 output columns
        v8f acc0 = {}, acc1 = {};
        for (int k0 = 0; k0 < EE; k0 += 32) {
            Frag16 a;   // 16-bit A 16x32: lane<16 K=0..15; lane>=16 K=16..31
            a.q[0] = *(const uint4*)(xs + lm * XPAD + k0 + lh * 16);
            a.q[1] = *(const uint4*)(xs + lm * XPAD + k0 + lh * 16 + 8);
            // B[k][n] = W[d+n][k0+k]: lane n fixed, K contiguous -> one v16h.
            v16h b0 = *(const v16h*)(W + (size_t)(d0 + lm) * EE + k0 + lh * 16);
            v16h b1 = *(const v16h*)(W + (size_t)(d0 + 16 + lm) * EE + k0 + lh * 16);
            acc0 = __builtin_amdgcn_wmma_f32_16x16x32_f16(
                false, a.v, false, b0, (short)0, acc0, false, false);
            acc1 = __builtin_amdgcn_wmma_f32_16x16x32_f16(
                false, a.v, false, b1, (short)0, acc1, false, false);
        }
        const float bias0 = bs[p][d0 + lm];
        const float bias1 = bs[p][d0 + 16 + lm];
        #pragma unroll
        for (int r = 0; r < 8; ++r) {   // D: VGPR r -> row r (+8 for hi lanes)
            oacc[(r + lh * 8) * OPAD + d0 + lm]      = acc0[r] + bias0;
            oacc[(r + lh * 8) * OPAD + d0 + 16 + lm] = acc1[r] + bias1;
        }
        __syncthreads();

        // LayerNorm over 256 columns per row: 16 threads/row partial sums.
        {
            const int r = tid >> 4, c16 = tid & 15;
            float s = 0.f, s2 = 0.f;
            #pragma unroll
            for (int j = 0; j < 16; ++j) {
                float v = oacc[r * OPAD + c16 * 16 + j];
                s += v; s2 += v * v;
            }
            ps[r][c16] = s; ps2[r][c16] = s2;
        }
        __syncthreads();
        if (tid < 16) {
            float s = 0.f, s2 = 0.f;
            #pragma unroll
            for (int j = 0; j < 16; ++j) { s += ps[tid][j]; s2 += ps2[tid][j]; }
            const float mu  = s * (1.0f / 256.0f);
            const float var = s2 * (1.0f / 256.0f) - mu * mu;
            mu_s[tid] = mu;
            rs_s[tid] = rsqrtf(var + 1e-5f);
        }
        __syncthreads();
        {
            const int r = tid >> 4, c16 = tid & 15;
            const float mu = mu_s[r], rv = rs_s[r];
            if (p < 2) {
                _Float16* dst = (p == 0 ? Qh : Kh)
                              + ((size_t)b * NN + n0 + r) * EE + c16 * 16;
                #pragma unroll
                for (int j = 0; j < 16; ++j)
                    dst[j] = (_Float16)((oacc[r * OPAD + c16 * 16 + j] - mu) * rv);
            } else {  // V transposed: Vt[b][e][m]
                #pragma unroll
                for (int j = 0; j < 16; ++j) {
                    const int e = c16 * 16 + j;
                    Vt[((size_t)b * EE + e) * NN + n0 + r] =
                        (_Float16)((oacc[r * OPAD + e] - mu) * rv);
                }
            }
        }
        __syncthreads();
    }
}

// ---------------------------------------------------------------------------
// Kernel 2: scores[b,n,m] = (Q[b,n,:] . K[b,m,:]) / 16.
// Block = 8 waves; each wave two 16x16 tiles (shared A-frag, independent
// accumulators); block covers 16n x 256m.
// B-frag = K^T column = K row -> contiguous 32B v16h load.
// ---------------------------------------------------------------------------
__global__ __launch_bounds__(256) void scores_kernel(
    const _Float16* __restrict__ Qh, const _Float16* __restrict__ Kh,
    float* __restrict__ scores)
{
    const int b     = blockIdx.x;        // 64
    const int n0    = blockIdx.y * 16;   // 32
    const int tid   = threadIdx.x;
    const int lane  = tid & 31;
    const int wave  = tid >> 5;
    const int lh    = lane >> 4;
    const int lm    = lane & 15;
    const int mbase = blockIdx.z * 256 + wave * 32;   // z: 2 chunks

    const _Float16* qrow  = Qh + ((size_t)b * NN + n0 + lm) * EE;
    const _Float16* krow0 = Kh + ((size_t)b * NN + mbase + lm) * EE;
    const _Float16* krow1 = krow0 + (size_t)16 * EE;

    v8f acc0 = {}, acc1 = {};
    for (int k0 = 0; k0 < EE; k0 += 32) {
        Frag16 a;
        a.q[0] = *(const uint4*)(qrow + k0 + lh * 16);
        a.q[1] = *(const uint4*)(qrow + k0 + lh * 16 + 8);
        v16h b0 = *(const v16h*)(krow0 + k0 + lh * 16);
        v16h b1 = *(const v16h*)(krow1 + k0 + lh * 16);
        acc0 = __builtin_amdgcn_wmma_f32_16x16x32_f16(
            false, a.v, false, b0, (short)0, acc0, false, false);
        acc1 = __builtin_amdgcn_wmma_f32_16x16x32_f16(
            false, a.v, false, b1, (short)0, acc1, false, false);
    }
    float* srow = scores + ((size_t)b * NN + n0) * NN + mbase + lm;
    #pragma unroll
    for (int r = 0; r < 8; ++r) {
        srow[(size_t)(r + lh * 8) * NN]      = acc0[r] * 0.0625f;  // 1/sqrt(256)
        srow[(size_t)(r + lh * 8) * NN + 16] = acc1[r] * 0.0625f;
    }
}

// ---------------------------------------------------------------------------
// Kernel 3: softmax over the BATCH axis. One thread per (n,m); the 64 batch
// values (1MB-strided) live in registers. Reads f32 scores once, writes the
// normalized attention weights as f16 (values in [0,1]) for the next GEMM.
// ---------------------------------------------------------------------------
__global__ __launch_bounds__(256) void softmax_batch_kernel(
    const float* __restrict__ scores, _Float16* __restrict__ attnh)
{
    const int idx = blockIdx.x * 256 + threadIdx.x;   // 0 .. 512*512-1
    const size_t stride = (size_t)NN * NN;
    float v[BB];
    float m = -3.402823466e38f;
    #pragma unroll
    for (int b = 0; b < BB; ++b) {
        v[b] = scores[(size_t)b * stride + idx];
        m = fmaxf(m, v[b]);
    }
    float s = 0.f;
    #pragma unroll
    for (int b = 0; b < BB; ++b) { v[b] = __expf(v[b] - m); s += v[b]; }
    const float inv = 1.0f / s;
    #pragma unroll
    for (int b = 0; b < BB; ++b)
        attnh[(size_t)b * stride + idx] = (_Float16)(v[b] * inv);
}

// ---------------------------------------------------------------------------
// Kernel 4: out[b,n,e] = sum_m attn[b,n,m] * V[b,m,e].
// attn already f16 -> A-frag is two 16B loads, no conversion. B-frag from
// Vt[b][e][m] rows (K-contiguous) -> single 32B v16h load. Each wave owns
// two e-tiles (shared A-frag); block covers 16n x 256e.
// ---------------------------------------------------------------------------
__global__ __launch_bounds__(256) void attnv_kernel(
    const _Float16* __restrict__ attnh, const _Float16* __restrict__ Vt,
    float* __restrict__ out)
{
    const int b     = blockIdx.x;        // 64
    const int n0    = blockIdx.y * 16;   // 32
    const int tid   = threadIdx.x;
    const int lane  = tid & 31;
    const int wave  = tid >> 5;
    const int lh    = lane >> 4;
    const int lm    = lane & 15;
    const int ebase = wave * 32;         // 8 waves cover all 256 e-columns

    const _Float16* arow  = attnh + ((size_t)b * NN + n0 + lm) * NN;
    const _Float16* vrow0 = Vt + ((size_t)b * EE + ebase + lm) * NN;
    const _Float16* vrow1 = vrow0 + (size_t)16 * NN;

    v8f acc0 = {}, acc1 = {};
    for (int k0 = 0; k0 < NN; k0 += 32) {
        Frag16 a;
        a.q[0] = *(const uint4*)(arow + k0 + lh * 16);
        a.q[1] = *(const uint4*)(arow + k0 + lh * 16 + 8);
        v16h b0 = *(const v16h*)(vrow0 + k0 + lh * 16);
        v16h b1 = *(const v16h*)(vrow1 + k0 + lh * 16);
        acc0 = __builtin_amdgcn_wmma_f32_16x16x32_f16(
            false, a.v, false, b0, (short)0, acc0, false, false);
        acc1 = __builtin_amdgcn_wmma_f32_16x16x32_f16(
            false, a.v, false, b1, (short)0, acc1, false, false);
    }
    float* orow = out + ((size_t)b * NN + n0) * EE + ebase + lm;
    #pragma unroll
    for (int r = 0; r < 8; ++r) {
        orow[(size_t)(r + lh * 8) * EE]      = acc0[r];
        orow[(size_t)(r + lh * 8) * EE + 16] = acc1[r];
    }
}

// ---------------------------------------------------------------------------
extern "C" void kernel_launch(void* const* d_in, const int* in_sizes, int n_in,
                              void* d_out, int out_size, void* d_ws, size_t ws_size,
                              hipStream_t stream)
{
    const float* x  = (const float*)d_in[0];
    const float* Wq = (const float*)d_in[1];
    const float* bq = (const float*)d_in[2];
    const float* Wk = (const float*)d_in[3];
    const float* bk = (const float*)d_in[4];
    const float* Wv = (const float*)d_in[5];
    const float* bv = (const float*)d_in[6];
    float* out = (float*)d_out;

    // Workspace: Wqh|Wkh|Wvh (f16, 128KB ea) | Qh|Kh|Vt (f16, 16.8MB ea)
    //          | scores (f32, 67MB) | attnh (f16, 33.5MB)
    char* ws = (char*)d_ws;
    const size_t w_bytes   = (size_t)EE * EE * sizeof(_Float16);
    const size_t qkv_bytes = (size_t)BB * NN * EE * sizeof(_Float16);
    _Float16* Wqh = (_Float16*)ws;  ws += w_bytes;
    _Float16* Wkh = (_Float16*)ws;  ws += w_bytes;
    _Float16* Wvh = (_Float16*)ws;  ws += w_bytes;
    _Float16* Qh  = (_Float16*)ws;  ws += qkv_bytes;
    _Float16* Kh  = (_Float16*)ws;  ws += qkv_bytes;
    _Float16* Vt  = (_Float16*)ws;  ws += qkv_bytes;
    float*    scores = (float*)ws;  ws += (size_t)BB * NN * NN * sizeof(float);
    _Float16* attnh  = (_Float16*)ws;

    cvt_weights_kernel<<<(EE * EE) / 256, 256, 0, stream>>>(Wq, Wk, Wv, Wqh, Wkh, Wvh);

    dim3 g1(BB, NN / 16);
    qkv_ln_kernel<<<g1, 256, 0, stream>>>(x, Wqh, bq, Wkh, bk, Wvh, bv, Qh, Kh, Vt);

    dim3 g2(BB, NN / 16, NN / 256);
    scores_kernel<<<g2, 256, 0, stream>>>(Qh, Kh, scores);

    softmax_batch_kernel<<<(NN * NN) / 256, 256, 0, stream>>>(scores, attnh);

    dim3 g4(BB, NN / 16);
    attnv_kernel<<<g4, 256, 0, stream>>>(attnh, Vt, out);
}
